// QMixer_57861799412372
// MI455X (gfx1250) — matrix operand order repaired and checked by reading, not compiled, not run on hardware
//
#include <hip/hip_runtime.h>

#define BB 128
#define NN 1024
#define FF 64

typedef __attribute__((ext_vector_type(16))) _Float16 v16h;
typedef __attribute__((ext_vector_type(8)))  _Float16 h8;
typedef __attribute__((ext_vector_type(8)))  float    v8f;
typedef __attribute__((ext_vector_type(4)))  float    f4;

struct WPtrs { const float* W[16]; };

struct MParams {
  const float* pair;
  const float* qv;
  const float* bias[16];
  const char*  wfrag;
  float* accW1; float* accB1; float* accWf; float* accV;
};

// B-matrix fragment (32x16 KxN, f16): 1KB per fragment, 32B per lane.
__device__ __forceinline__ v16h ldB(const char* wfrag, int mat16, int nt, int kt, int lane) {
  return *(const v16h*)(wfrag + ((((mat16 * 4 + nt) * 2) + kt) << 10) + (lane << 5));
}

// Build 16x32 f16 A fragment from a row-major f32 row (global or LDS).
// Lane layout: lanes 0-15 row M=lane hold K = kt*32 + {0..7, 16..23};
//              lanes 16-31 row M=lane-16 hold K = kt*32 + {8..15, 24..31}.
__device__ __forceinline__ v16h mk_afrag(const float* row, int kt, int ksel) {
  const float* p1 = row + kt * 32 + ksel * 8;
  f4 A0 = *(const f4*)(p1);
  f4 A1 = *(const f4*)(p1 + 4);
  f4 A2 = *(const f4*)(p1 + 16);
  f4 A3 = *(const f4*)(p1 + 20);
  v16h a;
#pragma unroll
  for (int i = 0; i < 4; ++i) {
    a[i]      = (_Float16)A0[i];
    a[4 + i]  = (_Float16)A1[i];
    a[8 + i]  = (_Float16)A2[i];
    a[12 + i] = (_Float16)A3[i];
  }
  return a;
}

__device__ __forceinline__ v8f wmma_f16(v16h a, v16h b, v8f c) {
  return __builtin_amdgcn_wmma_f32_16x16x32_f16(false, a, false, b, (short)0, c, false, false);
}

// Pre-convert the 16 f32 weight matrices into WMMA B-fragment f16 layout.
// frag id f = ((mat16*4 + nt)*2 + kt); B[k][n] = W[n][k] (x @ W.T).
__global__ __launch_bounds__(32) void qmix_prep(WPtrs wp, char* wfrag) {
  const int f = blockIdx.x;        // 0..127
  const int lane = threadIdx.x;    // 0..31
  const int mat16 = f >> 3;
  const int nt = (f >> 1) & 3;
  const int kt = f & 1;
  const float* W = wp.W[mat16];
  const int n  = lane & 15;
  const int kb = kt * 32 + (lane >> 4) * 8;
  v16h v;
#pragma unroll
  for (int j = 0; j < 16; ++j)
    v[j] = (_Float16)W[(nt * 16 + n) * FF + kb + (j < 8 ? j : j + 8)];
  *(v16h*)(wfrag + (f << 10) + (lane << 5)) = v;
}

__global__ __launch_bounds__(128) void qmix_main(MParams P) {
  __shared__ _Float16 sQKV[3][4][16 * FF];   // per-wave f16 Q/K/V tiles
  __shared__ float    sO[4][16 * FF];        // per-wave attention-out / mha-out tile
  __shared__ float    sAccW1[FF], sAccB1[FF], sAccWf[FF];
  __shared__ float    sAccV;
  __shared__ float    sRowScale[4][16], sRowMax[4][16];

  const int tid  = threadIdx.x;
  const int wave = tid >> 5;
  const int lane = tid & 31;
  const int b    = blockIdx.x >> 3;   // 8 workgroups per episode b
  const int sub  = blockIdx.x & 7;
  const int msel = lane & 15;
  const int ksel = lane >> 4;
  const int rlo  = ksel << 3;

  if (tid < FF) { sAccW1[tid] = 0.f; sAccB1[tid] = 0.f; sAccWf[tid] = 0.f; }
  if (tid == 0) sAccV = 0.f;
  __syncthreads();

  for (int t = 0; t < 2; ++t) {
    const int m0 = sub * 128 + t * 64 + wave * 16;   // agent row within b
    const int grow = b * NN + m0;
    const float* xrow = P.pair + (grow + msel) * FF;
    const v16h xa0 = mk_afrag(xrow, 0, ksel);
    const v16h xa1 = mk_afrag(xrow, 1, ksel);

    for (int net = 0; net < 4; ++net) {
      // ---- Q,K,V projections: [16,64] = [16,64] x [64,64], f16 WMMA ----
#pragma unroll
      for (int mat = 0; mat < 3; ++mat) {
        _Float16* dst = sQKV[mat][wave];
        const float* bvec = P.bias[net * 4 + mat];
#pragma unroll
        for (int nt = 0; nt < 4; ++nt) {
          v8f c = {};
          c = wmma_f16(xa0, ldB(P.wfrag, net * 4 + mat, nt, 0, lane), c);
          c = wmma_f16(xa1, ldB(P.wfrag, net * 4 + mat, nt, 1, lane), c);
          const int col = nt * 16 + msel;
          const float bia = bvec[col];
#pragma unroll
          for (int r = 0; r < 8; ++r)
            dst[(r + rlo) * FF + col] = (_Float16)(c[r] + bia);
        }
      }
      __builtin_amdgcn_wave_barrier();

      // ---- rank-1 attention: 16 rows x 8 heads = 128 tasks, 4 per lane ----
#pragma unroll
      for (int tk = 0; tk < 4; ++tk) {
        const int task = tk * 32 + lane;
        const int m = task >> 3, h = task & 7;
        h8 q8 = *(const h8*)&sQKV[0][wave][m * FF + h * 8];
        h8 k8 = *(const h8*)&sQKV[1][wave][m * FF + h * 8];
        h8 v8 = *(const h8*)&sQKV[2][wave][m * FF + h * 8];
        float qf[8], kf[8], vf[8];
#pragma unroll
        for (int i = 0; i < 8; ++i) { qf[i] = (float)q8[i]; kf[i] = (float)k8[i]; vf[i] = (float)v8[i]; }
        float oo[8];
#pragma unroll
        for (int i = 0; i < 8; ++i) {
          float mx = qf[i] * kf[0];
#pragma unroll
          for (int j = 1; j < 8; ++j) mx = fmaxf(mx, qf[i] * kf[j]);
          float s = 0.f, acc = 0.f;
#pragma unroll
          for (int j = 0; j < 8; ++j) {
            float e = __expf(qf[i] * kf[j] - mx);
            s += e; acc += e * vf[j];
          }
          oo[i] = acc / s;
        }
        f4 o0, o1;
#pragma unroll
        for (int i = 0; i < 4; ++i) { o0[i] = oo[i]; o1[i] = oo[4 + i]; }
        f4* od = (f4*)&sO[wave][m * FF + h * 8];
        od[0] = o0; od[1] = o1;
      }
      __builtin_amdgcn_wave_barrier();

      // ---- out = o @ Wo.T + bo ----
      {
        const float* orow = &sO[wave][msel * FF];
        const v16h oa0 = mk_afrag(orow, 0, ksel);
        const v16h oa1 = mk_afrag(orow, 1, ksel);
        const float* bo = P.bias[net * 4 + 3];
        v8f cw[4];
#pragma unroll
        for (int nt = 0; nt < 4; ++nt) {
          v8f c = {};
          c = wmma_f16(oa0, ldB(P.wfrag, net * 4 + 3, nt, 0, lane), c);
          c = wmma_f16(oa1, ldB(P.wfrag, net * 4 + 3, nt, 1, lane), c);
          cw[nt] = c;
        }
        __builtin_amdgcn_wave_barrier();
#pragma unroll
        for (int nt = 0; nt < 4; ++nt) {
          const int col = nt * 16 + msel;
          const float bia = bo[col];
#pragma unroll
          for (int r = 0; r < 8; ++r)
            sO[wave][(r + rlo) * FF + col] = cw[nt][r] + bia;
        }
      }
      __builtin_amdgcn_wave_barrier();

      // ---- per-net reduction into shared per-b accumulators ----
      if (net == 0) {
        // w1: row softmax over 64 feats, weighted by q_vals, summed over rows
        if (lane < 16) {
          const float* orow = &sO[wave][lane * FF];
          float mx = -1e30f;
          for (int i = 0; i < FF; i += 4) {
            f4 x = *(const f4*)(orow + i);
            mx = fmaxf(mx, fmaxf(fmaxf(x[0], x[1]), fmaxf(x[2], x[3])));
          }
          float s = 0.f;
          for (int i = 0; i < FF; i += 4) {
            f4 x = *(const f4*)(orow + i);
            s += __expf(x[0] - mx) + __expf(x[1] - mx) + __expf(x[2] - mx) + __expf(x[3] - mx);
          }
          sRowMax[wave][lane]   = mx;
          sRowScale[wave][lane] = P.qv[b * NN + m0 + lane] / s;
        }
        __builtin_amdgcn_wave_barrier();
#pragma unroll
        for (int cc = 0; cc < 2; ++cc) {
          const int e = lane + cc * 32;
          float acc = 0.f;
          for (int m = 0; m < 16; ++m)
            acc += sRowScale[wave][m] * __expf(sO[wave][m * FF + e] - sRowMax[wave][m]);
          atomicAdd(&sAccW1[e], acc);
        }
      } else if (net == 1 || net == 2) {
        float* tgt = (net == 1) ? sAccB1 : sAccWf;
#pragma unroll
        for (int cc = 0; cc < 2; ++cc) {
          const int e = lane + cc * 32;
          float s = 0.f;
          for (int m = 0; m < 16; ++m) s += sO[wave][m * FF + e];
          atomicAdd(&tgt[e], s);
        }
      } else {
        float s = 0.f;
        for (int cc = 0; cc < 2; ++cc) {
          const int e = lane + cc * 32;
          for (int m = 0; m < 16; ++m) s += sO[wave][m * FF + e];
        }
        atomicAdd(&sAccV, s);
      }
      __builtin_amdgcn_wave_barrier();
    }
  }

  __syncthreads();
  if (tid < FF) {
    atomicAdd(&P.accW1[b * FF + tid], sAccW1[tid]);
    atomicAdd(&P.accB1[b * FF + tid], sAccB1[tid]);
    atomicAdd(&P.accWf[b * FF + tid], sAccWf[tid]);
  }
  if (tid == 0) atomicAdd(&P.accV[b], sAccV);
}

__global__ __launch_bounds__(64) void qmix_final(const float* accW1, const float* accB1,
                                                 const float* accWf, const float* accV,
                                                 float* out) {
  __shared__ float sh[FF];
  const int e = threadIdx.x, b = blockIdx.x;
  const float invN = 1.0f / (float)NN;
  float hin = accW1[b * FF + e] + accB1[b * FF + e] * invN;
  float hidden = (hin > 0.f) ? hin : (__expf(hin) - 1.f);
  float wfv = accWf[b * FF + e] * invN;
  sh[e] = wfv; __syncthreads();
  float mx = sh[0];
  for (int i = 1; i < FF; ++i) mx = fmaxf(mx, sh[i]);
  float ex = __expf(wfv - mx);
  __syncthreads(); sh[e] = ex; __syncthreads();
  float s = 0.f;
  for (int i = 0; i < FF; ++i) s += sh[i];
  float part = hidden * (ex / s);
  __syncthreads(); sh[e] = part; __syncthreads();
  if (e == 0) {
    float q = 0.f;
    for (int i = 0; i < FF; ++i) q += sh[i];
    q += accV[b] * (1.0f / (float)(NN * FF));
    out[b] = q;
  }
}

extern "C" void kernel_launch(void* const* d_in, const int* in_sizes, int n_in,
                              void* d_out, int out_size, void* d_ws, size_t ws_size,
                              hipStream_t stream) {
  WPtrs wp; MParams P;
  P.pair = (const float*)d_in[0];
  P.qv   = (const float*)d_in[1];
  // dict order: for hn in {w1,b1,wf,v}: Wq,bq,Wk,bk,Wv,bv,Wo,bo
  for (int hn = 0; hn < 4; ++hn)
    for (int m = 0; m < 4; ++m) {
      wp.W[hn * 4 + m]   = (const float*)d_in[2 + hn * 8 + m * 2];
      P.bias[hn * 4 + m] = (const float*)d_in[2 + hn * 8 + m * 2 + 1];
    }
  char*  ws    = (char*)d_ws;
  char*  wfrag = ws;                           // 128 KB of f16 B-fragments
  float* acc   = (float*)(ws + 128 * 1024);    // per-b accumulators
  P.wfrag = wfrag;
  P.accW1 = acc;
  P.accB1 = acc + BB * FF;
  P.accWf = acc + 2 * BB * FF;
  P.accV  = acc + 3 * BB * FF;

  qmix_prep<<<128, 32, 0, stream>>>(wp, wfrag);
  hipMemsetAsync(acc, 0, (3 * BB * FF + BB) * sizeof(float), stream);
  qmix_main<<<1024, 128, 0, stream>>>(P);
  qmix_final<<<128, 64, 0, stream>>>(P.accW1, P.accB1, P.accWf, P.accV, (float*)d_out);
}
